// StructureEncoder_1675037245686
// MI455X (gfx1250) — 16-bit and 8-bit matrix operand order repaired and checked by reading, compiled, not run on hardware
//
#include <hip/hip_runtime.h>
#include <hip/hip_bf16.h>
#include <math.h>

#define NN 4096
#define EE 65536
#define HD 128
#define KP1 160
#define KP0 32
#define TK 64

typedef __bf16 bf16;
typedef __attribute__((ext_vector_type(16))) __bf16 v16bf;
typedef __attribute__((ext_vector_type(8)))  float  v8f;
typedef __attribute__((ext_vector_type(8)))  int    v8i;
typedef __attribute__((ext_vector_type(4)))  unsigned int u32x4;
typedef __attribute__((ext_vector_type(4)))  int    i32x4;
typedef __attribute__((ext_vector_type(8)))  int    i32x8;

#if defined(__has_builtin)
#if __has_builtin(__builtin_amdgcn_tensor_load_to_lds) && __has_builtin(__builtin_amdgcn_s_wait_tensorcnt)
#define HAVE_TDM 1
#endif
#endif
#ifndef HAVE_TDM
#define HAVE_TDM 0
#endif

static inline int cdiv(long a, long b) { return (int)((a + b - 1) / b); }

__device__ __forceinline__ unsigned fenc(float f) {
  unsigned u = __float_as_uint(f);
  return (u & 0x80000000u) ? ~u : (u | 0x80000000u);
}
__device__ __forceinline__ float fdec(unsigned u) {
  unsigned b = (u & 0x80000000u) ? (u ^ 0x80000000u) : ~u;
  return __uint_as_float(b);
}

__device__ __forceinline__ unsigned lds_off(const void* p) {
  return (unsigned)(unsigned long long)p;
}

#if HAVE_TDM
__device__ __forceinline__ void tdm_load_2d(unsigned ldsoff, const void* gptr,
                                            unsigned tile_w, unsigned tile_h,
                                            unsigned long long stride_elems,
                                            int ds_log2) {
  unsigned long long ga = (unsigned long long)gptr;
  u32x4 g0;
  g0[0] = 1u;
  g0[1] = ldsoff;
  g0[2] = (unsigned)(ga & 0xFFFFFFFFu);
  g0[3] = (unsigned)((ga >> 32) & 0x01FFFFFFu) | (2u << 30);
  i32x8 g1;
  g1[0] = (ds_log2 & 3) << 16;
  g1[1] = (int)((tile_w & 0xFFFFu) << 16);
  g1[2] = (int)(((tile_w >> 16) & 0xFFFFu) | ((tile_h & 0xFFFFu) << 16));
  g1[3] = (int)(((tile_h >> 16) & 0xFFFFu) | ((tile_w & 0xFFFFu) << 16));
  g1[4] = (int)(tile_h & 0xFFFFu);
  g1[5] = (int)(stride_elems & 0xFFFFFFFFull);
  g1[6] = (int)((stride_elems >> 32) & 0xFFFFull);
  g1[7] = 0;
  i32x4 z4 = {};
#if defined(__clang_major__) && (__clang_major__ >= 23)
  i32x8 z8 = {};
  __builtin_amdgcn_tensor_load_to_lds(g0, g1, z4, z4, z8, 0);
#else
  __builtin_amdgcn_tensor_load_to_lds(g0, g1, z4, z4, 0);
#endif
}
#endif

__global__ void fill_u32_k(unsigned* p, unsigned v, long n) {
  long i = (long)blockIdx.x * blockDim.x + threadIdx.x;
  if (i < n) p[i] = v;
}
__global__ void fill_f32_k(float* p, float v, long n) {
  long i = (long)blockIdx.x * blockDim.x + threadIdx.x;
  if (i < n) p[i] = v;
}
__global__ void copy_f_k(const float* s, float* d, long n) {
  long i = (long)blockIdx.x * blockDim.x + threadIdx.x;
  if (i < n) d[i] = s[i];
}
__global__ void copy_i_k(const int* s, int* d, long n) {
  long i = (long)blockIdx.x * blockDim.x + threadIdx.x;
  if (i < n) d[i] = s[i];
}

__global__ void deg_k(const int* src, float* deg, int E) {
  int e = blockIdx.x * blockDim.x + threadIdx.x;
  if (e < E) atomicAdd(&deg[src[e]], 1.0f);
}
__global__ void eattr0_k(const int* src, const int* dst, const float* pos,
                         float* eattr, int E) {
  int e = blockIdx.x * blockDim.x + threadIdx.x;
  if (e >= E) return;
  int s = src[e], d = dst[e];
#pragma unroll
  for (int f = 0; f < 3; ++f) eattr[e * 3 + f] = pos[d * 3 + f] - pos[s * 3 + f];
}
__global__ void write_level_k(const int* src, const int* dst, const int* emask,
                              const float* eattr, float* oe, float* oa, float* om, int E) {
  int e = blockIdx.x * blockDim.x + threadIdx.x;
  if (e >= E) return;
  oe[e] = (float)src[e];
  oe[E + e] = (float)dst[e];
#pragma unroll
  for (int f = 0; f < 3; ++f) oa[e * 3 + f] = eattr[e * 3 + f];
  om[e] = emask[e] ? 1.0f : 0.0f;
}

__global__ void pack_x_k(const float* feat, int dh, const float* pos,
                         bf16* X, int n, int Kp) {
  int idx = blockIdx.x * blockDim.x + threadIdx.x;
  if (idx >= n * Kp) return;
  int i = idx / Kp, k = idx % Kp;
  float v = 0.0f;
  if (k < dh) v = feat[(long)i * dh + k];
  else if (k < dh + 3) v = pos[i * 3 + (k - dh)];
  X[idx] = (bf16)v;
}
__global__ void pack_w_k(const float* W, bf16* Wt, int d, int Kp) {
  int idx = blockIdx.x * blockDim.x + threadIdx.x;
  if (idx >= Kp * HD) return;
  int h = idx / Kp, k = idx % Kp;
  Wt[idx] = (k < d) ? (bf16)W[k * HD + h] : (bf16)0.0f;
}

__global__ __launch_bounds__(128) void gemm_bf16_k(const bf16* __restrict__ X,
                                                   const bf16* __restrict__ Wt,
                                                   float* __restrict__ C, int Kp) {
  int tM = blockIdx.x;
  int wave = threadIdx.x >> 5, lane = threadIdx.x & 31;
  int tN = blockIdx.y * 4 + wave;
  int half = lane >> 4, mn = lane & 15;
  __shared__ uint4 As4[64];
  __shared__ uint4 Bs4[4][64];
  const bf16* As = (const bf16*)As4;
  const bf16* Bt = (const bf16*)Bs4[wave];
  v8f acc = {};
  for (int k0 = 0; k0 < Kp; k0 += 32) {
#if HAVE_TDM
    if (wave == 0)
      tdm_load_2d(lds_off(As4), X + (size_t)(tM * 16) * Kp + k0, 32, 16,
                  (unsigned long long)Kp,  1);
    tdm_load_2d(lds_off(Bs4[wave]), Wt + (size_t)(tN * 16) * Kp + k0, 32, 16,
                (unsigned long long)Kp,  1);
    __builtin_amdgcn_s_wait_tensorcnt(0);
    __syncthreads();
#else
    if (wave == 0)
      for (int i = lane; i < 64; i += 32) {
        int r = i >> 2, q = i & 3;
        As4[i] = *(const uint4*)&X[(size_t)(tM * 16 + r) * Kp + k0 + q * 8];
      }
    for (int i = lane; i < 64; i += 32) {
      int r = i >> 2, q = i & 3;
      Bs4[wave][i] = *(const uint4*)&Wt[(size_t)(tN * 16 + r) * Kp + k0 + q * 8];
    }
    __syncthreads();
#endif
    union { uint4 q[2]; v16bf v; } au, bu;
    au.q[0] = *(const uint4*)&As[mn * 32 + half * 8];
    au.q[1] = *(const uint4*)&As[mn * 32 + 16 + half * 8];
    bu.q[0] = *(const uint4*)&Bt[mn * 32 + half * 8];
    bu.q[1] = *(const uint4*)&Bt[mn * 32 + 16 + half * 8];
    acc = __builtin_amdgcn_wmma_f32_16x16x32_bf16(false, au.v, false, bu.v,
                                                  (short)0, acc, false, false);
    __syncthreads();
  }
#pragma unroll
  for (int r = 0; r < 8; ++r) {
    int gm = tM * 16 + r + half * 8;
    C[(long)gm * HD + tN * 16 + mn] = acc[r];
  }
}

__global__ void loop_acc_k(const int* dst, const int* emask, const float* eattr,
                           float* sum3, float* cnt, int E) {
  int e = blockIdx.x * blockDim.x + threadIdx.x;
  if (e >= E || !emask[e]) return;
  int d = dst[e];
#pragma unroll
  for (int f = 0; f < 3; ++f) atomicAdd(&sum3[d * 3 + f], eattr[e * 3 + f]);
  atomicAdd(&cnt[d], 1.0f);
}
__global__ void loop_div_k(float* sum3, const float* cnt, int n) {
  int i = blockIdx.x * blockDim.x + threadIdx.x;
  if (i >= n) return;
  float c = fmaxf(cnt[i], 1.0f);
#pragma unroll
  for (int f = 0; f < 3; ++f) sum3[i * 3 + f] /= c;
}
__global__ void logit_k(const float* hl, const float* hr, const float* We,
                        const float* att, const float* eattr, const float* loop_attr,
                        const int* src, const int* dst, const int* emask,
                        float* logit, unsigned* mxenc, int E, int n) {
  int e = blockIdx.x * blockDim.x + threadIdx.x;
  if (e >= E + n) return;
  int s, d, m; float ea0, ea1, ea2;
  if (e < E) {
    s = src[e]; d = dst[e]; m = emask[e];
    ea0 = eattr[e * 3]; ea1 = eattr[e * 3 + 1]; ea2 = eattr[e * 3 + 2];
  } else {
    s = d = e - E; m = 1;
    ea0 = loop_attr[s * 3]; ea1 = loop_attr[s * 3 + 1]; ea2 = loop_attr[s * 3 + 2];
  }
  if (!m) { logit[e] = -INFINITY; return; }
  float acc = 0.0f;
  for (int h = 0; h < HD; ++h) {
    float g = hl[(long)s * HD + h] + hr[(long)d * HD + h]
            + ea0 * We[h] + ea1 * We[HD + h] + ea2 * We[2 * HD + h];
    g = (g > 0.0f) ? g : 0.2f * g;
    acc += g * att[h];
  }
  logit[e] = acc;
  atomicMax(&mxenc[d], fenc(acc));
}
__global__ void ex_k(const float* logit, const unsigned* mxenc,
                     const int* src, const int* dst, const int* emask,
                     float* exv, float* sumex, int E, int n) {
  int e = blockIdx.x * blockDim.x + threadIdx.x;
  if (e >= E + n) return;
  int d, m;
  if (e < E) { d = dst[e]; m = emask[e]; } else { d = e - E; m = 1; }
  float v = 0.0f;
  if (m) { v = __expf(logit[e] - fdec(mxenc[d])); atomicAdd(&sumex[d], v); }
  exv[e] = v;
}
__global__ void agg_k(const float* hl, const float* exv, const float* sumex,
                      const int* src, const int* dst, const int* emask,
                      float* out, int E) {
  int e = blockIdx.x, h = threadIdx.x;
  int s, d, m;
  if (e < E) { s = src[e]; d = dst[e]; m = emask[e]; } else { s = d = e - E; m = 1; }
  if (!m) return;
  float a = exv[e] / sumex[d];
  atomicAdd(&out[(long)d * HD + h], a * hl[(long)s * HD + h]);
}
__global__ void finalize_k(const float* agg, const float* b, float* hout, long n) {
  long i = (long)blockIdx.x * blockDim.x + threadIdx.x;
  if (i >= n) return;
  float v = agg[i] + b[i % HD];
  hout[i] = (v > 0.0f) ? v : (__expf(v) - 1.0f);
}

__global__ void tvec_k(const float* h, const float* pos, const float* w_rel,
                       const float* w_root, const float* b_rel,
                       float* tvec, float* svec, int n) {
  int i = blockIdx.x * blockDim.x + threadIdx.x;
  if (i >= n) return;
  float t = 0.0f, r = 0.0f;
  for (int f = 0; f < HD; ++f) { float v = h[(long)i * HD + f]; t += v * w_rel[f]; r += v * w_root[f]; }
#pragma unroll
  for (int f = 0; f < 3; ++f) { float v = pos[i * 3 + f]; t += v * w_rel[HD + f]; r += v * w_root[HD + f]; }
  tvec[i] = t; svec[i] = b_rel[0] + r;
}
__global__ void adj_fill_k(const int* src, const int* dst, const int* emask,
                           unsigned char* adj, unsigned char* adjT, int n, int E) {
  int e = blockIdx.x * blockDim.x + threadIdx.x;
  if (e < E && emask[e]) {
    adj[(long)src[e] * n + dst[e]] = 1;
    adjT[(long)dst[e] * n + src[e]] = 1;
  }
}

__global__ __launch_bounds__(128) void adj_square_score_k(const unsigned char* __restrict__ adj,
                                                          const unsigned char* __restrict__ adjT,
                                                          const float* __restrict__ t,
                                                          float* __restrict__ svec, int n) {
  int tileJ = blockIdx.x;
  int wave = threadIdx.x >> 5, lane = threadIdx.x & 31;
  int tileI = blockIdx.y * 4 + wave;
  int half = lane >> 4, mn = lane & 15;
  __shared__ uint4 As4[64];
  __shared__ uint4 Bs4[4][64];
  const unsigned* Asw = (const unsigned*)As4;
  const unsigned* Btw = (const unsigned*)Bs4[wave];
  v8i acc = {};
  for (int k0 = 0; k0 < n; k0 += TK) {
#if HAVE_TDM
    if (wave == 0)
      tdm_load_2d(lds_off(As4), adj + (size_t)(tileJ * 16) * n + k0, 64, 16,
                  (unsigned long long)n,  0);
    tdm_load_2d(lds_off(Bs4[wave]), adjT + (size_t)(tileI * 16) * n + k0, 64, 16,
                (unsigned long long)n,  0);
    __builtin_amdgcn_s_wait_tensorcnt(0);
    __syncthreads();
#else
    if (wave == 0)
      for (int i = lane; i < 64; i += 32) {
        int r = i >> 2, q = i & 3;
        As4[i] = *(const uint4*)&adj[(size_t)(tileJ * 16 + r) * n + k0 + q * 16];
      }
    for (int i = lane; i < 64; i += 32) {
      int r = i >> 2, q = i & 3;
      Bs4[wave][i] = *(const uint4*)&adjT[(size_t)(tileI * 16 + r) * n + k0 + q * 16];
    }
    __syncthreads();
#endif
    v8i a, b;
#pragma unroll
    for (int v = 0; v < 8; ++v) {
      int ak = ((v >> 2) * 32) + (((v >> 1) & 1) * 16) + half * 8 + ((v & 1) * 4);
      a[v] = (int)Asw[(mn * TK + ak) >> 2];
      int bk = ((v >> 2) * 32) + (((v >> 1) & 1) * 16) + half * 8 + ((v & 1) * 4);
      b[v] = (int)Btw[(mn * TK + bk) >> 2];
    }
    acc = __builtin_amdgcn_wmma_i32_16x16x64_iu8(false, a, false, b, acc, false, false);
    __syncthreads();
  }
  int gcol = tileI * 16 + mn;
  float partial = 0.0f;
#pragma unroll
  for (int r = 0; r < 8; ++r) {
    int grow = tileJ * 16 + r + half * 8;
    int aug = (acc[r] > 0) || (adj[(long)grow * n + gcol] != 0);
    partial += aug ? t[grow] : 0.0f;
  }
  __shared__ float colsum[4][16];
  if (half == 1) colsum[wave][mn] = partial;
  __syncthreads();
  if (half == 0) atomicAdd(&svec[gcol], partial + colsum[wave][mn]);
}

__global__ void tanh_k(const float* svec, float* score, int n) {
  int i = blockIdx.x * blockDim.x + threadIdx.x;
  if (i < n) score[i] = tanhf(svec[i]);
}
__global__ void select_k(const float* score, const float* h, const float* pos,
                         float* hnew, float* posnew, float* outP, int* newidx,
                         int n, int k) {
  int i = blockIdx.x * blockDim.x + threadIdx.x;
  if (i >= n) return;
  float si = score[i];
  int rank = 0;
  for (int j = 0; j < n; ++j) {
    float sj = score[j];
    rank += (sj > si) || (sj == si && j < i);
  }
  if (rank < k) {
    newidx[i] = rank;
    for (int f = 0; f < HD; ++f)
      hnew[(long)rank * HD + f] = h[(long)i * HD + f] * si;
#pragma unroll
    for (int f = 0; f < 3; ++f) {
      float p = pos[i * 3 + f];
      posnew[rank * 3 + f] = p;
      outP[rank * 3 + f] = p;
    }
  } else {
    newidx[i] = -1;
  }
}
__global__ void remap_k(const int* newidx, const float* posnew,
                        int* src, int* dst, int* emask, float* eattr, int E) {
  int e = blockIdx.x * blockDim.x + threadIdx.x;
  if (e >= E) return;
  int ns = newidx[src[e]], nd = newidx[dst[e]];
  int valid = emask[e] && (ns >= 0) && (nd >= 0);
  ns = valid ? ns : 0; nd = valid ? nd : 0;
  src[e] = ns; dst[e] = nd; emask[e] = valid;
#pragma unroll
  for (int f = 0; f < 3; ++f)
    eattr[e * 3 + f] = valid ? (posnew[nd * 3 + f] - posnew[ns * 3 + f]) : 0.0f;
}

struct ConvP { const float *Wl, *Wr, *We, *att, *b; };

static void run_conv(const ConvP& P, const float* feat, int dh, int Kp,
                     const float* pos, int n,
                     const int* src, const int* dst, const int* emask, const float* eattr,
                     bf16* xbf, bf16* wlb, bf16* wrb,
                     float* hl, float* hr, float* aggbuf,
                     float* sum3, float* cnt, unsigned* mxenc, float* sumex,
                     float* logit, float* exv, float* hout, hipStream_t st) {
  int d = dh + 3;
  pack_x_k<<<cdiv((long)n * Kp, 256), 256, 0, st>>>(feat, dh, pos, xbf, n, Kp);
  pack_w_k<<<cdiv(Kp * HD, 256), 256, 0, st>>>(P.Wl, wlb, d, Kp);
  pack_w_k<<<cdiv(Kp * HD, 256), 256, 0, st>>>(P.Wr, wrb, d, Kp);
  dim3 gg(n / 16, HD / 64);
  gemm_bf16_k<<<gg, 128, 0, st>>>(xbf, wlb, hl, Kp);
  gemm_bf16_k<<<gg, 128, 0, st>>>(xbf, wrb, hr, Kp);
  fill_f32_k<<<cdiv(n * 3, 256), 256, 0, st>>>(sum3, 0.0f, n * 3);
  fill_f32_k<<<cdiv(n, 256), 256, 0, st>>>(cnt, 0.0f, n);
  loop_acc_k<<<cdiv(EE, 256), 256, 0, st>>>(dst, emask, eattr, sum3, cnt, EE);
  loop_div_k<<<cdiv(n, 256), 256, 0, st>>>(sum3, cnt, n);
  fill_u32_k<<<cdiv(n, 256), 256, 0, st>>>(mxenc, 0x007FFFFFu, n);
  logit_k<<<cdiv(EE + n, 256), 256, 0, st>>>(hl, hr, P.We, P.att, eattr, sum3,
                                             src, dst, emask, logit, mxenc, EE, n);
  fill_f32_k<<<cdiv(n, 256), 256, 0, st>>>(sumex, 0.0f, n);
  ex_k<<<cdiv(EE + n, 256), 256, 0, st>>>(logit, mxenc, src, dst, emask, exv, sumex, EE, n);
  fill_f32_k<<<cdiv((long)n * HD, 256), 256, 0, st>>>(aggbuf, 0.0f, (long)n * HD);
  agg_k<<<EE + n, HD, 0, st>>>(hl, exv, sumex, src, dst, emask, aggbuf, EE);
  finalize_k<<<cdiv((long)n * HD, 256), 256, 0, st>>>(aggbuf, P.b, hout, (long)n * HD);
}

static void run_pool(const float* w_rel, const float* w_root, const float* b_rel,
                     const float* hcur, const float* pos, int n, int k,
                     unsigned char* adj, unsigned char* adjT,
                     float* tvec, float* svec, float* score,
                     int* newidx, float* hnew, float* posnew,
                     int* src, int* dst, int* emask, float* eattr,
                     float* outE, float* outP, float* outA, float* outM,
                     hipStream_t st) {
  tvec_k<<<cdiv(n, 128), 128, 0, st>>>(hcur, pos, w_rel, w_root, b_rel, tvec, svec, n);
  fill_u32_k<<<cdiv((long)n * n / 4, 256), 256, 0, st>>>((unsigned*)adj, 0u, (long)n * n / 4);
  fill_u32_k<<<cdiv((long)n * n / 4, 256), 256, 0, st>>>((unsigned*)adjT, 0u, (long)n * n / 4);
  adj_fill_k<<<cdiv(EE, 256), 256, 0, st>>>(src, dst, emask, adj, adjT, n, EE);
  dim3 gt(n / 16, n / 64);
  adj_square_score_k<<<gt, 128, 0, st>>>(adj, adjT, tvec, svec, n);
  tanh_k<<<cdiv(n, 256), 256, 0, st>>>(svec, score, n);
  select_k<<<cdiv(n, 128), 128, 0, st>>>(score, hcur, pos, hnew, posnew, outP, newidx, n, k);
  remap_k<<<cdiv(EE, 256), 256, 0, st>>>(newidx, posnew, src, dst, emask, eattr, EE);
  write_level_k<<<cdiv(EE, 256), 256, 0, st>>>(src, dst, emask, eattr, outE, outA, outM, EE);
}

extern "C" void kernel_launch(void* const* d_in, const int* in_sizes, int n_in,
                              void* d_out, int out_size, void* d_ws, size_t ws_size,
                              hipStream_t stream) {
  (void)in_sizes; (void)n_in; (void)out_size; (void)ws_size;
  const int*   ei   = (const int*)d_in[1];
  const float* pos0 = (const float*)d_in[2];
  ConvP conv[4];
  for (int c = 0; c < 4; ++c) {
    conv[c].Wl  = (const float*)d_in[3 + 5 * c + 0];
    conv[c].Wr  = (const float*)d_in[3 + 5 * c + 1];
    conv[c].We  = (const float*)d_in[3 + 5 * c + 2];
    conv[c].att = (const float*)d_in[3 + 5 * c + 3];
    conv[c].b   = (const float*)d_in[3 + 5 * c + 4];
  }
  const float* p_wrel[2]  = { (const float*)d_in[23], (const float*)d_in[26] };
  const float* p_wroot[2] = { (const float*)d_in[24], (const float*)d_in[27] };
  const float* p_brel[2]  = { (const float*)d_in[25], (const float*)d_in[28] };

  char* w = (char*)d_ws; size_t off = 0;
  auto alloc = [&](size_t bytes) -> void* {
    off = (off + 255) & ~(size_t)255; void* p = w + off; off += bytes; return p;
  };
  unsigned char* adj  = (unsigned char*)alloc((size_t)NN * NN);
  unsigned char* adjT = (unsigned char*)alloc((size_t)NN * NN);
  bf16* xbf = (bf16*)alloc((size_t)NN * KP1 * sizeof(bf16));
  bf16* wlb = (bf16*)alloc((size_t)KP1 * HD * sizeof(bf16));
  bf16* wrb = (bf16*)alloc((size_t)KP1 * HD * sizeof(bf16));
  float* hl     = (float*)alloc((size_t)NN * HD * 4);
  float* hr     = (float*)alloc((size_t)NN * HD * 4);
  float* aggbuf = (float*)alloc((size_t)NN * HD * 4);
  float* hA     = (float*)alloc((size_t)NN * HD * 4);
  float* hB     = (float*)alloc((size_t)NN * HD * 4);
  float* posA   = (float*)alloc((size_t)NN * 3 * 4);
  float* posB   = (float*)alloc((size_t)NN * 3 * 4);
  int* srcC  = (int*)alloc((size_t)EE * 4);
  int* dstC  = (int*)alloc((size_t)EE * 4);
  int* maskC = (int*)alloc((size_t)EE * 4);
  float* eattrC = (float*)alloc((size_t)EE * 3 * 4);
  float* logit  = (float*)alloc((size_t)(EE + NN) * 4);
  float* exv    = (float*)alloc((size_t)(EE + NN) * 4);
  unsigned* mxenc = (unsigned*)alloc((size_t)NN * 4);
  float* sumex = (float*)alloc((size_t)NN * 4);
  float* sum3  = (float*)alloc((size_t)NN * 3 * 4);
  float* cnt   = (float*)alloc((size_t)NN * 4);
  float* deg   = (float*)alloc((size_t)NN * 4);
  float* tvec  = (float*)alloc((size_t)NN * 4);
  float* svec  = (float*)alloc((size_t)NN * 4);
  float* score = (float*)alloc((size_t)NN * 4);
  int* newidx  = (int*)alloc((size_t)NN * 4);

  float* out = (float*)d_out;
  const size_t O_E2 = 0,       O_E1 = 131072,  O_E0 = 262144;
  const size_t O_P2 = 393216,  O_P1 = 396288,  O_P0 = 402432;
  const size_t O_A2 = 414720,  O_A1 = 611328,  O_A0 = 807936;
  const size_t O_M2 = 1004544, O_M1 = 1070080, O_M0 = 1135616;

  copy_i_k<<<cdiv(EE, 256), 256, 0, stream>>>(ei, srcC, EE);
  copy_i_k<<<cdiv(EE, 256), 256, 0, stream>>>(ei + EE, dstC, EE);
  fill_u32_k<<<cdiv(EE, 256), 256, 0, stream>>>((unsigned*)maskC, 1u, EE);
  copy_f_k<<<cdiv(NN * 3, 256), 256, 0, stream>>>(pos0, posA, NN * 3);
  fill_f32_k<<<cdiv(NN, 256), 256, 0, stream>>>(deg, 0.0f, NN);
  deg_k<<<cdiv(EE, 256), 256, 0, stream>>>(srcC, deg, EE);
  eattr0_k<<<cdiv(EE, 256), 256, 0, stream>>>(srcC, dstC, pos0, eattrC, EE);

  write_level_k<<<cdiv(EE, 256), 256, 0, stream>>>(srcC, dstC, maskC, eattrC,
                                                   out + O_E0, out + O_A0, out + O_M0, EE);
  copy_f_k<<<cdiv(NN * 3, 256), 256, 0, stream>>>(pos0, out + O_P0, NN * 3);

  run_conv(conv[0], deg, 1, KP0, posA, NN, srcC, dstC, maskC, eattrC,
           xbf, wlb, wrb, hl, hr, aggbuf, sum3, cnt, mxenc, sumex, logit, exv, hA, stream);
  run_conv(conv[1], hA, HD, KP1, posA, NN, srcC, dstC, maskC, eattrC,
           xbf, wlb, wrb, hl, hr, aggbuf, sum3, cnt, mxenc, sumex, logit, exv, hB, stream);

  run_pool(p_wrel[0], p_wroot[0], p_brel[0], hB, posA, NN, 2048,
           adj, adjT, tvec, svec, score, newidx, hA, posB, srcC, dstC, maskC, eattrC,
           out + O_E1, out + O_P1, out + O_A1, out + O_M1, stream);

  run_conv(conv[2], hA, HD, KP1, posB, 2048, srcC, dstC, maskC, eattrC,
           xbf, wlb, wrb, hl, hr, aggbuf, sum3, cnt, mxenc, sumex, logit, exv, hB, stream);

  run_pool(p_wrel[1], p_wroot[1], p_brel[1], hB, posB, 2048, 1024,
           adj, adjT, tvec, svec, score, newidx, hA, posA, srcC, dstC, maskC, eattrC,
           out + O_E2, out + O_P2, out + O_A2, out + O_M2, stream);

}
